// MultiHeadAttention_14516989460940
// MI455X (gfx1250) — matrix operand order repaired and checked by reading, not compiled, not run on hardware
//
#include <hip/hip_runtime.h>

// ---------------------------------------------------------------------------
// MI455X (gfx1250, wave32) multi-head attention forward.
// All GEMM work on v_wmma_f32_16x16x32_f16 (f16 operands, f32 accumulate).
// Compute-bound: ~51 GFLOP dense vs ~60MB HBM (=> ~2.6us at 23.3 TB/s);
// K/V (17MB) live in the 192MB L2, so flash re-reads are L2 hits.
// This revision software-pipelines all WMMA loops (register double-buffering)
// so fragment loads for step i+1 are in flight while step i's WMMAs execute —
// the round-0 assembly showed s_wait_loadcnt 0x0 in front of every WMMA.
// ---------------------------------------------------------------------------

typedef __attribute__((ext_vector_type(16))) _Float16 v16h;
typedef __attribute__((ext_vector_type(8)))  _Float16 v8h;
typedef __attribute__((ext_vector_type(8)))  float    v8f;

#define WMMA16x16x32(a, b, c) \
  __builtin_amdgcn_wmma_f32_16x16x32_f16(false, (a), false, (b), (short)0, (c), false, false)

namespace {
constexpr int BS = 4, SLEN = 2048, DIM = 512, H = 8, DPH = 64;
constexpr int NTOK = BS * SLEN;                 // 8192
constexpr size_t XH_ELTS = (size_t)NTOK * DIM;  // 4,194,304 (= B*H*S*DPH too)
constexpr size_t W_ELTS  = (size_t)DIM * DIM;   // 262,144

// workspace layout (bytes, all offsets 256-aligned)
constexpr size_t OFF_XH  = 0;
constexpr size_t OFF_WQ  = OFF_XH + XH_ELTS * 2;
constexpr size_t OFF_WK  = OFF_WQ + W_ELTS * 2;
constexpr size_t OFF_WV  = OFF_WK + W_ELTS * 2;
constexpr size_t OFF_WO  = OFF_WV + W_ELTS * 2;
constexpr size_t OFF_Q   = OFF_WO + W_ELTS * 2;   // f16 [B,H,S,DPH], pre-scaled 1/8
constexpr size_t OFF_K   = OFF_Q  + XH_ELTS * 2;  // f16 [B,H,S,DPH]
constexpr size_t OFF_VT  = OFF_K  + XH_ELTS * 2;  // f16 [B,H,DPH,S]  (V transposed)
constexpr size_t OFF_CTX = OFF_VT + XH_ELTS * 2;  // f16 [B,S,DIM]
}  // namespace

// ---------------------------------------------------------------------------
// A-fragment loader: 16x32 f16 tile, row-major source with leading dim `ld`.
// Contiguous k order used here (load_bT matches): lane L holds row M=L%16; half element e maps to
// K = (L/16)*16 + e  => two contiguous 16B chunks per lane.
// ---------------------------------------------------------------------------
__device__ __forceinline__ v16h load_a16x32(const _Float16* __restrict__ tile,
                                            int ld, int lane) {
  const int l16 = lane & 15, hh = (lane >> 4) & 1;
  const _Float16* rp = tile + (size_t)l16 * ld + hh * 16;
  v8h lo = *(const v8h*)(rp);
  v8h hi = *(const v8h*)(rp + 8);
  v16h a;
#pragma unroll
  for (int i = 0; i < 8; ++i) { a[i] = lo[i]; a[i + 8] = hi[i]; }
  return a;
}

// ---------------------------------------------------------------------------
// B-fragment loader for B = M^T where M is row-major with leading dim `ld`:
// B[k][n] = M[n][k]. Lane L supplies column n = L%16 with 16 contiguous
// halfs starting at k = (L/16)*16  => one 32B contiguous load per lane.
// ---------------------------------------------------------------------------
__device__ __forceinline__ v16h load_bT(const _Float16* __restrict__ m_n0_k0,
                                        int ld, int lane) {
  const int l16 = lane & 15, hh = (lane >> 4) & 1;
  return *(const v16h*)(m_n0_k0 + (size_t)l16 * ld + hh * 16);
}

// ---------------------------------------------------------------------------
// Kernel 1: fp32 -> f16 conversion of x and the four weight matrices.
// ---------------------------------------------------------------------------
__global__ void cvt_f32_to_f16_kernel(const float* __restrict__ x,
                                      const float* __restrict__ wq,
                                      const float* __restrict__ wk,
                                      const float* __restrict__ wv,
                                      const float* __restrict__ wo,
                                      _Float16* __restrict__ xh,
                                      _Float16* __restrict__ wqh,
                                      _Float16* __restrict__ wkh,
                                      _Float16* __restrict__ wvh,
                                      _Float16* __restrict__ woh) {
  const size_t i = (size_t)blockIdx.x * blockDim.x + threadIdx.x;
  const size_t stride = (size_t)gridDim.x * blockDim.x;
  for (size_t t = i; t < XH_ELTS; t += stride) xh[t] = (_Float16)x[t];
  for (size_t t = i; t < W_ELTS; t += stride) {
    wqh[t] = (_Float16)wq[t];
    wkh[t] = (_Float16)wk[t];
    wvh[t] = (_Float16)wv[t];
    woh[t] = (_Float16)wo[t];
  }
}

// ---------------------------------------------------------------------------
// Pipelined 32x64 GEMM tile: acc[rt][nt] += Arow(rt) * W^T(nt), K = DIM.
// Double-buffered fragments: loads for step i+1 are issued before step i's
// WMMAs so the matrix pipe overlaps memory (waits become loadcnt > 0).
// ---------------------------------------------------------------------------
__device__ __forceinline__ void gemm32x64_pipelined(
    const _Float16* __restrict__ abase,   // &A[row0][0], ld = DIM
    const _Float16* __restrict__ wbase,   // &W[nb][0],  ld = DIM
    int lane, v8f acc[2][4]) {
  v16h a0 = load_a16x32(abase, DIM, lane);
  v16h a1 = load_a16x32(abase + 16 * DIM, DIM, lane);
  v16h b0 = load_bT(wbase + 0 * 16 * DIM, DIM, lane);
  v16h b1 = load_bT(wbase + 1 * 16 * DIM, DIM, lane);
  v16h b2 = load_bT(wbase + 2 * 16 * DIM, DIM, lane);
  v16h b3 = load_bT(wbase + 3 * 16 * DIM, DIM, lane);

  for (int k0 = 32; k0 < DIM; k0 += 32) {
    // issue next-step loads first
    v16h na0 = load_a16x32(abase + k0, DIM, lane);
    v16h na1 = load_a16x32(abase + 16 * DIM + k0, DIM, lane);
    v16h nb0 = load_bT(wbase + 0 * 16 * DIM + k0, DIM, lane);
    v16h nb1 = load_bT(wbase + 1 * 16 * DIM + k0, DIM, lane);
    v16h nb2 = load_bT(wbase + 2 * 16 * DIM + k0, DIM, lane);
    v16h nb3 = load_bT(wbase + 3 * 16 * DIM + k0, DIM, lane);
    // consume previous-step fragments (already resident)
    acc[0][0] = WMMA16x16x32(a0, b0, acc[0][0]);
    acc[0][1] = WMMA16x16x32(a0, b1, acc[0][1]);
    acc[0][2] = WMMA16x16x32(a0, b2, acc[0][2]);
    acc[0][3] = WMMA16x16x32(a0, b3, acc[0][3]);
    acc[1][0] = WMMA16x16x32(a1, b0, acc[1][0]);
    acc[1][1] = WMMA16x16x32(a1, b1, acc[1][1]);
    acc[1][2] = WMMA16x16x32(a1, b2, acc[1][2]);
    acc[1][3] = WMMA16x16x32(a1, b3, acc[1][3]);
    a0 = na0; a1 = na1; b0 = nb0; b1 = nb1; b2 = nb2; b3 = nb3;
  }
  acc[0][0] = WMMA16x16x32(a0, b0, acc[0][0]);
  acc[0][1] = WMMA16x16x32(a0, b1, acc[0][1]);
  acc[0][2] = WMMA16x16x32(a0, b2, acc[0][2]);
  acc[0][3] = WMMA16x16x32(a0, b3, acc[0][3]);
  acc[1][0] = WMMA16x16x32(a1, b0, acc[1][0]);
  acc[1][1] = WMMA16x16x32(a1, b1, acc[1][1]);
  acc[1][2] = WMMA16x16x32(a1, b2, acc[1][2]);
  acc[1][3] = WMMA16x16x32(a1, b3, acc[1][3]);
}

// ---------------------------------------------------------------------------
// Kernel 2: fused QKV projection. One wave per 32x64 output tile.
// blockIdx.y: [0,24): /8 selects Q|K|V, %8 selects 64-col group.
// Q pre-scaled by 1/sqrt(64); K row-major per head; V stored transposed.
// ---------------------------------------------------------------------------
__global__ void __launch_bounds__(32)
qkv_proj_kernel(const _Float16* __restrict__ xh,
                const _Float16* __restrict__ wqh, const float* __restrict__ bq,
                const _Float16* __restrict__ wkh, const float* __restrict__ bk,
                const _Float16* __restrict__ wvh, const float* __restrict__ bv,
                _Float16* __restrict__ Qo, _Float16* __restrict__ Ko,
                _Float16* __restrict__ Vt) {
  const int lane = threadIdx.x;
  const int l16 = lane & 15, hh = (lane >> 4) & 1;
  const int row0 = blockIdx.x * 32;
  const int which = blockIdx.y >> 3;          // 0=Q 1=K 2=V
  const int nb = (blockIdx.y & 7) * 64;

  const _Float16* w = (which == 0) ? wqh : (which == 1) ? wkh : wvh;
  const float* bias = (which == 0) ? bq  : (which == 1) ? bk  : bv;

  v8f acc[2][4] = {{{}, {}, {}, {}}, {{}, {}, {}, {}}};
  gemm32x64_pipelined(xh + (size_t)row0 * DIM, w + (size_t)nb * DIM, lane, acc);

  const float qscale = (which == 0) ? 0.125f : 1.0f;
#pragma unroll
  for (int rt = 0; rt < 2; ++rt) {
#pragma unroll
    for (int nt = 0; nt < 4; ++nt) {
      const int j = nb + nt * 16 + l16;   // feature 0..511
      const int hd = j >> 6, d = j & 63;  // head, dim-in-head
      const float bj = bias[j];
#pragma unroll
      for (int r = 0; r < 8; ++r) {
        const int t = row0 + rt * 16 + r + hh * 8;  // token
        const int b_ = t >> 11, s = t & (SLEN - 1);
        const float v = (acc[rt][nt][r] + bj) * qscale;
        if (which == 0)
          Qo[(((size_t)b_ * H + hd) * SLEN + s) * DPH + d] = (_Float16)v;
        else if (which == 1)
          Ko[(((size_t)b_ * H + hd) * SLEN + s) * DPH + d] = (_Float16)v;
        else
          Vt[(((size_t)b_ * H + hd) * DPH + d) * SLEN + s] = (_Float16)v;
      }
    }
  }
}

// ---------------------------------------------------------------------------
// Kernel 3: causal flash attention. One wave per (b,h, 16-query tile).
// Pipelining: V fragments for tile kt are issued BEFORE the score WMMAs
// (consumed only after softmax, whose VALU work hides their latency), and
// K fragments for tile kt+1 are issued before the PV WMMAs, so the score
// WMMAs always consume already-resident registers.
// P (C/D layout) -> LDS (16x32, stride 40 halfs) -> A-fragment reload;
// same-wave DS ops are in-order so no barrier is needed (single-wave WG).
// ---------------------------------------------------------------------------
__global__ void __launch_bounds__(32)
flash_attn_kernel(const _Float16* __restrict__ Q,
                  const _Float16* __restrict__ K,
                  const _Float16* __restrict__ Vt,
                  _Float16* __restrict__ ctxh) {
  const int lane = threadIdx.x;
  const int l16 = lane & 15, hh = (lane >> 4) & 1;
  const int bh = blockIdx.y;
  const int b = bh >> 3, h = bh & 7;
  const int qb = blockIdx.x * 16;

  __shared__ _Float16 lds_p[16 * 40];  // padded stride: 80B, 16B-aligned rows

  const _Float16* Qb = Q + (size_t)bh * SLEN * DPH;
  const _Float16* Kb = K + (size_t)bh * SLEN * DPH;
  const _Float16* Vb = Vt + (size_t)bh * DPH * SLEN;

  // Q tile 16x64 as two K=32 A-fragments, resident for the whole kernel.
  const v16h aq0 = load_a16x32(Qb + (size_t)qb * DPH + 0,  DPH, lane);
  const v16h aq1 = load_a16x32(Qb + (size_t)qb * DPH + 32, DPH, lane);

  v8f acc[4] = {{}, {}, {}, {}};
  float m[8], l[8];
#pragma unroll
  for (int r = 0; r < 8; ++r) { m[r] = -1e30f; l[r] = 0.0f; }

  // K B-fragment pointers: B[d][n] = K[n][d]; index [s(16-key subtile)][d half]
  const _Float16* kr0 = Kb + (size_t)l16 * DPH + hh * 16;   // subtile s=0
  const _Float16* kr1 = kr0 + 16 * DPH;                     // subtile s=1

  // preload K fragments for kt = 0
  v16h bk00 = *(const v16h*)(kr0);
  v16h bk01 = *(const v16h*)(kr0 + 32);
  v16h bk10 = *(const v16h*)(kr1);
  v16h bk11 = *(const v16h*)(kr1 + 32);

  const int nkt = (qb + 16 + 31) >> 5;  // 32-key tiles up to the diagonal
  for (int kt = 0; kt < nkt; ++kt) {
    const int kb = kt << 5;

    // Issue V loads now; consumed only after softmax.
    v16h bv[4];
#pragma unroll
    for (int nt = 0; nt < 4; ++nt)
      bv[nt] = *(const v16h*)(Vb + (size_t)(nt * 16 + l16) * SLEN + kb + hh * 16);

    // S = Q * K^T  (two 16x16 col-subtiles, inner dim 64 = 2 WMMAs each);
    // K fragments already resident from the previous iteration.
    v8f sc[2];
    {
      v8f t0 = {};
      t0 = WMMA16x16x32(aq0, bk00, t0);
      t0 = WMMA16x16x32(aq1, bk01, t0);
      sc[0] = t0;
      v8f t1 = {};
      t1 = WMMA16x16x32(aq0, bk10, t1);
      t1 = WMMA16x16x32(aq1, bk11, t1);
      sc[1] = t1;
    }

    // Preload next tile's K fragments (overlaps softmax + PV).
    if (kt + 1 < nkt) {
      const size_t off = (size_t)(kb + 32) * DPH;
      bk00 = *(const v16h*)(kr0 + off);
      bk01 = *(const v16h*)(kr0 + off + 32);
      bk10 = *(const v16h*)(kr1 + off);
      bk11 = *(const v16h*)(kr1 + off + 32);
    }

    if (kb + 31 > qb) {  // diagonal-crossing tile: elementwise causal mask
#pragma unroll
      for (int s = 0; s < 2; ++s) {
        const int kcol = kb + s * 16 + l16;
#pragma unroll
        for (int r = 0; r < 8; ++r)
          if (kcol > qb + r + hh * 8) sc[s][r] = -1e30f;
      }
    }

    // Online softmax. Row of element r is (r + hh*8); its 16 columns live in
    // one 16-lane half => xor-butterfly 1/2/4/8 stays inside the half.
#pragma unroll
    for (int r = 0; r < 8; ++r) {
      float mt = fmaxf(sc[0][r], sc[1][r]);
#pragma unroll
      for (int off = 1; off < 16; off <<= 1)
        mt = fmaxf(mt, __shfl_xor(mt, off, 32));
      const float mn = fmaxf(m[r], mt);
      const float rescale = __expf(m[r] - mn);
      m[r] = mn;
      const float p0 = __expf(sc[0][r] - mn);
      const float p1 = __expf(sc[1][r] - mn);
      float ps = p0 + p1;
#pragma unroll
      for (int off = 1; off < 16; off <<= 1)
        ps += __shfl_xor(ps, off, 32);
      l[r] = l[r] * rescale + ps;
#pragma unroll
      for (int nt = 0; nt < 4; ++nt) acc[nt][r] *= rescale;
      lds_p[(r + hh * 8) * 40 + l16]      = (_Float16)p0;
      lds_p[(r + hh * 8) * 40 + 16 + l16] = (_Float16)p1;
    }

    // Reload P in A-fragment layout (same-wave DS ops are in-order).
    v16h ap;
    {
      const _Float16* pr = &lds_p[l16 * 40 + hh * 16];
      const v8h plo = *(const v8h*)(pr);
      const v8h phi = *(const v8h*)(pr + 8);
#pragma unroll
      for (int i = 0; i < 8; ++i) { ap[i] = plo[i]; ap[i + 8] = phi[i]; }
    }

    // ctx += P * V   (B from V^T rows; loads were issued before the scores)
#pragma unroll
    for (int nt = 0; nt < 4; ++nt)
      acc[nt] = WMMA16x16x32(ap, bv[nt], acc[nt]);
  }

  // Normalize and store ctx as f16 [B,S,DIM] for the output projection.
  float inv[8];
#pragma unroll
  for (int r = 0; r < 8; ++r) inv[r] = 1.0f / l[r];
#pragma unroll
  for (int nt = 0; nt < 4; ++nt) {
    const int col = h * DPH + nt * 16 + l16;
#pragma unroll
    for (int r = 0; r < 8; ++r) {
      const int srow = qb + r + hh * 8;
      ctxh[((size_t)b * SLEN + srow) * DIM + col] =
          (_Float16)(acc[nt][r] * inv[r]);
    }
  }
}

// ---------------------------------------------------------------------------
// Kernel 4: output projection  out = ctx @ wo^T + bo  (fp32 result).
// ---------------------------------------------------------------------------
__global__ void __launch_bounds__(32)
out_proj_kernel(const _Float16* __restrict__ ctxh,
                const _Float16* __restrict__ woh,
                const float* __restrict__ bo,
                float* __restrict__ out) {
  const int lane = threadIdx.x;
  const int l16 = lane & 15, hh = (lane >> 4) & 1;
  const int row0 = blockIdx.x * 32;
  const int nb = blockIdx.y * 64;

  v8f acc[2][4] = {{{}, {}, {}, {}}, {{}, {}, {}, {}}};
  gemm32x64_pipelined(ctxh + (size_t)row0 * DIM, woh + (size_t)nb * DIM, lane,
                      acc);

#pragma unroll
  for (int rt = 0; rt < 2; ++rt) {
#pragma unroll
    for (int nt = 0; nt < 4; ++nt) {
      const int j = nb + nt * 16 + l16;
      const float bj = bo[j];
#pragma unroll
      for (int r = 0; r < 8; ++r) {
        const int t = row0 + rt * 16 + r + hh * 8;
        out[(size_t)t * DIM + j] = acc[rt][nt][r] + bj;
      }
    }
  }
}

// ---------------------------------------------------------------------------
extern "C" void kernel_launch(void* const* d_in, const int* in_sizes, int n_in,
                              void* d_out, int out_size, void* d_ws,
                              size_t ws_size, hipStream_t stream) {
  (void)in_sizes; (void)n_in; (void)out_size; (void)ws_size;

  const float* x  = (const float*)d_in[0];
  // d_in[1] = mask: causal, recomputed analytically on-device (unused).
  const float* wq = (const float*)d_in[2];
  const float* bq = (const float*)d_in[3];
  const float* wk = (const float*)d_in[4];
  const float* bk = (const float*)d_in[5];
  const float* wv = (const float*)d_in[6];
  const float* bv = (const float*)d_in[7];
  const float* wo = (const float*)d_in[8];
  const float* bo = (const float*)d_in[9];
  float* out = (float*)d_out;

  char* ws = (char*)d_ws;
  _Float16* xh  = (_Float16*)(ws + OFF_XH);
  _Float16* wqh = (_Float16*)(ws + OFF_WQ);
  _Float16* wkh = (_Float16*)(ws + OFF_WK);
  _Float16* wvh = (_Float16*)(ws + OFF_WV);
  _Float16* woh = (_Float16*)(ws + OFF_WO);
  _Float16* Qh  = (_Float16*)(ws + OFF_Q);
  _Float16* Kh  = (_Float16*)(ws + OFF_K);
  _Float16* Vth = (_Float16*)(ws + OFF_VT);
  _Float16* ctx = (_Float16*)(ws + OFF_CTX);

  cvt_f32_to_f16_kernel<<<4096, 256, 0, stream>>>(x, wq, wk, wv, wo, xh, wqh,
                                                  wkh, wvh, woh);

  qkv_proj_kernel<<<dim3(NTOK / 32, 24), 32, 0, stream>>>(
      xh, wqh, bq, wkh, bk, wvh, bv, Qh, Kh, Vth);

  flash_attn_kernel<<<dim3(SLEN / 16, BS * H), 32, 0, stream>>>(Qh, Kh, Vth,
                                                                ctx);

  out_proj_kernel<<<dim3(NTOK / 32, DIM / 64), 32, 0, stream>>>(ctx, woh, bo,
                                                                out);
}
